// MHPLSTMCore_2181843386922
// MI455X (gfx1250) — compile-verified, matrix operand order checked by reading, not run on hardware
//
#include <hip/hip_runtime.h>

// ---------------------------------------------------------------------------
// MHPLSTM core for MI455X (gfx1250, wave32, WMMA f16 16x16x32).
// B=4, S=2048, NH=8, AD=128, OD=128, FH=512.
// GEMM: block tile 128x128, 8 waves, wave tile 32x64, double-buffered LDS.
// Weights pre-converted to f16 and pre-transposed col-major once per launch.
// Cell recurrence via 3-pass chunked parallel scan (16 chunks x 128).
// ---------------------------------------------------------------------------

typedef __attribute__((ext_vector_type(16))) _Float16 v16h;
typedef __attribute__((ext_vector_type(8)))  _Float16 v8h;
typedef __attribute__((ext_vector_type(8)))  float    v8f;

#define NH   8
#define AD   128
#define OD   128
#define FH   512
#define SEQ  2048
#define BSZ  4
#define NTOK (BSZ * SEQ)   // 8192 rows
#define NCH  16            // scan chunks
#define CLEN (SEQ / NCH)   // 128

union F16Frag { v16h v; v8h h[2]; };

// Load a 16-half WMMA fragment from LDS for this lane.
// ISA layout (16-bit A/B 16x32): lane<16 holds K{0..7,16..23} of row (lane),
// lane>=16 holds K{8..15,24..31} of row (lane-16).  p points at (row, k=0).
__device__ __forceinline__ v16h ld_frag(const _Float16* p, int hl) {
    F16Frag f;
    f.h[0] = *reinterpret_cast<const v8h*>(p + hl * 8);
    f.h[1] = *reinterpret_cast<const v8h*>(p + 16 + hl * 8);
    return f.v;
}

// Block tile 128(M) x 128(N); 8 waves; wave tile 32(M) x 64(N).
// A: f16 activations, row stride lda halves.
// Bt: f16 weights, col-major [col][K] (row stride = K halves).
// As/Bs: double-buffered LDS, 2 * 4096 halves each.
template <int K>
__device__ __forceinline__ void gemm_tile(
    const _Float16* __restrict__ Abase, int lda,
    const _Float16* __restrict__ Bt,
    _Float16* As, _Float16* Bs, v8f acc[2][4])
{
    const int tid  = threadIdx.x;
    const int lane = tid & 31;
    const int wave = tid >> 5;
    const int wm   = (wave & 3) * 32;   // wave M offset
    const int wn   = (wave >> 2) * 64;  // wave N offset
    const int hl   = lane >> 4;
    const int lr   = lane & 15;

    // Per-thread staging addresses (2 x 16B chunks of the 128x32 tile each).
    const int c0 = tid, c1 = tid + 256;
    const int r0 = c0 >> 2, k0 = (c0 & 3) * 8;
    const int r1 = c1 >> 2, k1 = (c1 & 3) * 8;
    const _Float16* Ap0 = Abase + (size_t)r0 * lda + k0;
    const _Float16* Ap1 = Abase + (size_t)r1 * lda + k1;
    const _Float16* Bp0 = Bt + (size_t)r0 * K + k0;
    const _Float16* Bp1 = Bt + (size_t)r1 * K + k1;
    const int d0 = r0 * 32 + k0, d1 = r1 * 32 + k1;

    auto stage = [&](int kt, int buf) {
        _Float16* a = As + buf * 4096;
        _Float16* b = Bs + buf * 4096;
        *reinterpret_cast<v8h*>(&a[d0]) = *reinterpret_cast<const v8h*>(Ap0 + kt);
        *reinterpret_cast<v8h*>(&a[d1]) = *reinterpret_cast<const v8h*>(Ap1 + kt);
        *reinterpret_cast<v8h*>(&b[d0]) = *reinterpret_cast<const v8h*>(Bp0 + kt);
        *reinterpret_cast<v8h*>(&b[d1]) = *reinterpret_cast<const v8h*>(Bp1 + kt);
    };

    stage(0, 0);
    int buf = 0;
    for (int kt = 0; kt < K; kt += 32) {
        __syncthreads();
        if (kt + 32 < K) stage(kt + 32, buf ^ 1);  // overlap with tile-t compute

        const _Float16* a = As + buf * 4096;
        const _Float16* b = Bs + buf * 4096;
        v16h afrag[2], bfrag[4];
#pragma unroll
        for (int i = 0; i < 2; ++i)
            afrag[i] = ld_frag(&a[(wm + i * 16 + lr) * 32], hl);
#pragma unroll
        for (int j = 0; j < 4; ++j)
            bfrag[j] = ld_frag(&b[(wn + j * 16 + lr) * 32], hl);

#pragma unroll
        for (int i = 0; i < 2; ++i)
#pragma unroll
            for (int j = 0; j < 4; ++j)
                acc[i][j] = __builtin_amdgcn_wmma_f32_16x16x32_f16(
                    false, afrag[i], false, bfrag[j],
                    (short)0, acc[i][j], false, false);
        buf ^= 1;
    }
}

__device__ __forceinline__ float sigmoidf_(float v) {
    return 1.0f / (1.0f + __expf(-v));
}

// ---------------------------------------------------------------------------
// Weight pre-transpose: src [NH][K][N] f32 row-major ->
// dst [NH][NTOT][K] f16 col-major at column offset c0.
// ---------------------------------------------------------------------------
__global__ void wtrans_kernel(const float* __restrict__ src,
                              _Float16* __restrict__ dst,
                              int K, int N, int NTOT, int c0)
{
    const int n = blockIdx.y;
    int e = blockIdx.x * 256 + threadIdx.x;     // over K*N
    int k = e / N;
    int c = e - k * N;
    dst[((size_t)n * NTOT + c0 + c) * K + k] =
        (_Float16)src[((size_t)n * K + k) * N + c];
}

// ---------------------------------------------------------------------------
// Kernel 1: shifted cumulative sum along S. csum[b,s,n,d] = sum_{t<s} x[b,t,n,d]
// ---------------------------------------------------------------------------
__global__ void cumsum_kernel(const float* __restrict__ x, float* __restrict__ csum)
{
    int b = blockIdx.x >> 3;
    int n = blockIdx.x & 7;
    int d = threadIdx.x;
    float acc = 0.0f;
    size_t idx = (((size_t)b * SEQ) * NH + n) * AD + d;
    const size_t stride = (size_t)NH * AD;
#pragma unroll 4
    for (int s = 0; s < SEQ; ++s) {
        csum[idx] = acc;
        acc += x[idx];
        idx += stride;
    }
}

// ---------------------------------------------------------------------------
// Kernel 2: LayerNorm over (NH*AD)=1024 channels per (b,s); build f16 gh =
// [x | LN(csum)] and the x-half of f16 og_in.
// ---------------------------------------------------------------------------
__global__ void __launch_bounds__(256) ln_build_kernel(
    const float* __restrict__ x, const float* __restrict__ csum,
    const float* __restrict__ ln_w, const float* __restrict__ ln_b,
    _Float16* __restrict__ gh, _Float16* __restrict__ ogin)
{
    const int row = blockIdx.x;           // 0..NTOK-1
    const int tid = threadIdx.x;
    float v[4], sum = 0.f, ss = 0.f;
#pragma unroll
    for (int j = 0; j < 4; ++j) {
        int e = tid + j * 256;
        float c = csum[(size_t)row * 1024 + e];
        v[j] = c; sum += c; ss += c * c;
    }
#pragma unroll
    for (int off = 16; off > 0; off >>= 1) {
        sum += __shfl_down(sum, off);
        ss  += __shfl_down(ss,  off);
    }
    __shared__ float rs[8], rq[8];
    __shared__ float s_mean, s_rstd;
    int lane = tid & 31, wave = tid >> 5;
    if (lane == 0) { rs[wave] = sum; rq[wave] = ss; }
    __syncthreads();
    if (tid == 0) {
        float S1 = 0.f, S2 = 0.f;
#pragma unroll
        for (int w = 0; w < 8; ++w) { S1 += rs[w]; S2 += rq[w]; }
        float mean = S1 * (1.0f / 1024.0f);
        float var  = S2 * (1.0f / 1024.0f) - mean * mean;
        s_mean = mean;
        s_rstd = rsqrtf(var + 1e-6f);
    }
    __syncthreads();
    float mean = s_mean, rstd = s_rstd;
#pragma unroll
    for (int j = 0; j < 4; ++j) {
        int e = tid + j * 256;
        int n = e >> 7, d = e & 127;
        float xv = x[(size_t)row * 1024 + e];
        float cn = (v[j] - mean) * rstd * ln_w[e] + ln_b[e];
        size_t gbase = ((size_t)row * NH + n) * 256;
        gh[gbase + d]        = (_Float16)xv;
        gh[gbase + 128 + d]  = (_Float16)cn;
        ogin[gbase + d]      = (_Float16)xv;
    }
}

// ---------------------------------------------------------------------------
// Kernel 3: Y = gh @ [W_ifg | W_h1]  (K=256, N=768 per head), 128-col tiles:
//   ct==0 -> igate = sigmoid ; ct==1 -> fgate = sigmoid ; ct>=2 -> h = relu f16
// Uniform branch hoisted OUTSIDE the store nests (no per-element branching).
// ---------------------------------------------------------------------------
__global__ void __launch_bounds__(256) gemm1_kernel(
    const _Float16* __restrict__ gh, const _Float16* __restrict__ wt1,
    const float* __restrict__ b_ifg, const float* __restrict__ b_h1,
    float* __restrict__ fgate, float* __restrict__ igate,
    _Float16* __restrict__ hbuf)
{
    __shared__ __align__(16) _Float16 As[2 * 4096];
    __shared__ __align__(16) _Float16 Bs[2 * 4096];
    const int rt = blockIdx.x;   // 128-row tile
    const int ct = blockIdx.y;   // 128-col tile of 768
    const int n  = blockIdx.z;

    const _Float16* Abase = gh  + ((size_t)rt * 128 * NH + n) * 256;
    const _Float16* Bt    = wt1 + ((size_t)n * 768 + ct * 128) * 256;

    v8f acc[2][4] = {};
    gemm_tile<256>(Abase, NH * 256, Bt, As, Bs, acc);

    const int lane = threadIdx.x & 31, wave = threadIdx.x >> 5;
    const int wm = (wave & 3) * 32, wn = (wave >> 2) * 64;
    const int hl = lane >> 4, lr = lane & 15;
    const int row0 = rt * 128 + wm + hl * 8;

    if (ct == 0) {
        const float* bias = b_ifg + n * 256;
#pragma unroll
        for (int i = 0; i < 2; ++i)
#pragma unroll
            for (int j = 0; j < 4; ++j) {
                int colL = wn + j * 16 + lr;
                float bv = bias[colL];
#pragma unroll
                for (int r = 0; r < 8; ++r) {
                    size_t base = (size_t)(row0 + i * 16 + r) * NH + n;
                    igate[base * 128 + colL] = sigmoidf_(acc[i][j][r] + bv);
                }
            }
    } else if (ct == 1) {
        const float* bias = b_ifg + n * 256 + 128;
#pragma unroll
        for (int i = 0; i < 2; ++i)
#pragma unroll
            for (int j = 0; j < 4; ++j) {
                int colL = wn + j * 16 + lr;
                float bv = bias[colL];
#pragma unroll
                for (int r = 0; r < 8; ++r) {
                    size_t base = (size_t)(row0 + i * 16 + r) * NH + n;
                    fgate[base * 128 + colL] = sigmoidf_(acc[i][j][r] + bv);
                }
            }
    } else {
        const int cbase = (ct - 2) * 128;
        const float* bias = b_h1 + n * 512 + cbase;
#pragma unroll
        for (int i = 0; i < 2; ++i)
#pragma unroll
            for (int j = 0; j < 4; ++j) {
                int colL = wn + j * 16 + lr;
                float bv = bias[colL];
#pragma unroll
                for (int r = 0; r < 8; ++r) {
                    size_t base = (size_t)(row0 + i * 16 + r) * NH + n;
                    hbuf[base * 512 + cbase + colL] =
                        (_Float16)fmaxf(acc[i][j][r] + bv, 0.f);
                }
            }
    }
}

// ---------------------------------------------------------------------------
// Kernel 4: hidden = h @ W_h2 + b_h2 ; igh = igate * hidden (in-place)
// ---------------------------------------------------------------------------
__global__ void __launch_bounds__(256) gemm2_kernel(
    const _Float16* __restrict__ hbuf, const _Float16* __restrict__ wt2,
    const float* __restrict__ b_h2, float* __restrict__ igate_igh)
{
    __shared__ __align__(16) _Float16 As[2 * 4096];
    __shared__ __align__(16) _Float16 Bs[2 * 4096];
    const int rt = blockIdx.x;
    const int n  = blockIdx.z;

    const _Float16* Abase = hbuf + ((size_t)rt * 128 * NH + n) * 512;
    const _Float16* Bt    = wt2 + (size_t)n * 128 * 512;
    const float* bias     = b_h2 + n * 128;

    v8f acc[2][4] = {};
    gemm_tile<512>(Abase, NH * 512, Bt, As, Bs, acc);

    const int lane = threadIdx.x & 31, wave = threadIdx.x >> 5;
    const int wm = (wave & 3) * 32, wn = (wave >> 2) * 64;
    const int hl = lane >> 4, lr = lane & 15;
    const int row0 = rt * 128 + wm + hl * 8;
#pragma unroll
    for (int i = 0; i < 2; ++i)
#pragma unroll
        for (int j = 0; j < 4; ++j) {
            int colL = wn + j * 16 + lr;
            float bv = bias[colL];
#pragma unroll
            for (int r = 0; r < 8; ++r) {
                size_t idx = ((size_t)(row0 + i * 16 + r) * NH + n) * 128 + colL;
                float hidden = acc[i][j][r] + bv;
                igate_igh[idx] = igate_igh[idx] * hidden;   // igh = igate*hidden
            }
        }
}

// ---------------------------------------------------------------------------
// Kernels 5a/5b/5c: chunked parallel scan for c[s] = f[s]*c[s-1] + igh[s].
// 5a: per-chunk composition (A = prod f, B = local scan from 0).
// 5b: 16-step sequential carry scan per (b,n,d), c[-1] = init_cx.
// 5c: apply carry, write f32 cell and f16 cell-half of og_in.
// ---------------------------------------------------------------------------
__global__ void scan_partial_kernel(
    const float* __restrict__ fgate, const float* __restrict__ igh,
    float* __restrict__ Aout, float* __restrict__ Bout)
{
    int chunk = blockIdx.x & (NCH - 1);
    int bn    = blockIdx.x >> 4;          // b*NH+n, 0..31
    int b = bn >> 3, n = bn & 7;
    int d = threadIdx.x;
    size_t idx = (((size_t)b * SEQ + chunk * CLEN) * NH + n) * 128 + d;
    const size_t stride = (size_t)NH * 128;
    float a = 1.0f, bv = 0.0f;
#pragma unroll 4
    for (int s = 0; s < CLEN; ++s) {
        float f = fgate[idx], g = igh[idx];
        a *= f;
        bv = fmaf(f, bv, g);
        idx += stride;
    }
    size_t o = ((size_t)bn * NCH + chunk) * 128 + d;
    Aout[o] = a;
    Bout[o] = bv;
}

__global__ void scan_carry_kernel(
    const float* __restrict__ Ain, const float* __restrict__ Bin,
    float* __restrict__ carry, const float* __restrict__ init_cx)
{
    int bn = blockIdx.x;                  // 0..31
    int n  = bn & 7;
    int d  = threadIdx.x;
    float c = init_cx[n * 128 + d];
#pragma unroll
    for (int j = 0; j < NCH; ++j) {
        size_t o = ((size_t)bn * NCH + j) * 128 + d;
        carry[o] = c;                     // carry-in for chunk j
        c = fmaf(Ain[o], c, Bin[o]);
    }
}

__global__ void scan_apply_kernel(
    const float* __restrict__ fgate, const float* __restrict__ igh,
    const float* __restrict__ carry,
    float* __restrict__ cell, _Float16* __restrict__ ogin)
{
    int chunk = blockIdx.x & (NCH - 1);
    int bn    = blockIdx.x >> 4;
    int b = bn >> 3, n = bn & 7;
    int d = threadIdx.x;
    float c = carry[((size_t)bn * NCH + chunk) * 128 + d];
    size_t idx  = (((size_t)b * SEQ + chunk * CLEN) * NH + n) * 128 + d;
    size_t gidx = (((size_t)b * SEQ + chunk * CLEN) * NH + n) * 256 + 128 + d;
    const size_t stride  = (size_t)NH * 128;
    const size_t gstride = (size_t)NH * 256;
#pragma unroll 4
    for (int s = 0; s < CLEN; ++s) {
        c = fmaf(fgate[idx], c, igh[idx]);
        cell[idx] = c;
        ogin[gidx] = (_Float16)c;
        idx += stride;
        gidx += gstride;
    }
}

// ---------------------------------------------------------------------------
// Kernel 6: ogate = sigmoid(og_in @ W_og + b_og) ; out = ogate * cell
// ---------------------------------------------------------------------------
__global__ void __launch_bounds__(256) gemm3_kernel(
    const _Float16* __restrict__ ogin, const _Float16* __restrict__ wt3,
    const float* __restrict__ b_og, const float* __restrict__ cell,
    float* __restrict__ out)
{
    __shared__ __align__(16) _Float16 As[2 * 4096];
    __shared__ __align__(16) _Float16 Bs[2 * 4096];
    const int rt = blockIdx.x;
    const int n  = blockIdx.z;

    const _Float16* Abase = ogin + ((size_t)rt * 128 * NH + n) * 256;
    const _Float16* Bt    = wt3 + (size_t)n * 128 * 256;
    const float* bias     = b_og + n * 128;

    v8f acc[2][4] = {};
    gemm_tile<256>(Abase, NH * 256, Bt, As, Bs, acc);

    const int lane = threadIdx.x & 31, wave = threadIdx.x >> 5;
    const int wm = (wave & 3) * 32, wn = (wave >> 2) * 64;
    const int hl = lane >> 4, lr = lane & 15;
    const int row0 = rt * 128 + wm + hl * 8;
#pragma unroll
    for (int i = 0; i < 2; ++i)
#pragma unroll
        for (int j = 0; j < 4; ++j) {
            int colL = wn + j * 16 + lr;
            float bv = bias[colL];
#pragma unroll
            for (int r = 0; r < 8; ++r) {
                size_t idx = ((size_t)(row0 + i * 16 + r) * NH + n) * 128 + colL;
                float og = sigmoidf_(acc[i][j][r] + bv);
                out[idx] = og * cell[idx];
            }
        }
}

// ---------------------------------------------------------------------------
// Host launcher.  Workspace layout (all offsets 16B aligned, ~201 MB):
//   seg0 [0,32M)     : csum f32 (k1->k2)   then fgate f32 (gemm1->scan)
//   seg1 [32M,64M)   : gh f16   (k2->gemm1) then cell f32 (scan->gemm3)
//   seg2 [64M,96M)   : og_in f16 (k2,scan -> gemm3)
//   seg3 [96M,128M)  : igate f32 (gemm1->gemm2) then igh (gemm2->scan)
//   seg4 [128M,192M) : h f16 (gemm1->gemm2)
//   [192M,...)       : f16 weights wt1/wt2/wt3 (~4.7 MB), scan temporaries
// ---------------------------------------------------------------------------
extern "C" void kernel_launch(void* const* d_in, const int* in_sizes, int n_in,
                              void* d_out, int out_size, void* d_ws, size_t ws_size,
                              hipStream_t stream) {
    const float* x       = (const float*)d_in[0];
    const float* W_ifg   = (const float*)d_in[1];
    const float* b_ifg   = (const float*)d_in[2];
    const float* W_h1    = (const float*)d_in[3];
    const float* b_h1    = (const float*)d_in[4];
    const float* W_h2    = (const float*)d_in[5];
    const float* b_h2    = (const float*)d_in[6];
    const float* W_og    = (const float*)d_in[7];
    const float* b_og    = (const float*)d_in[8];
    const float* ln_w    = (const float*)d_in[9];
    const float* ln_b    = (const float*)d_in[10];
    const float* init_cx = (const float*)d_in[11];
    float* out = (float*)d_out;

    const size_t SEG = 33554432ull;   // 32 MiB
    char* ws = (char*)d_ws;
    float*    csum  = (float*)(ws + 0 * SEG);
    float*    fgate = (float*)(ws + 0 * SEG);   // aliases csum (dead after k2)
    _Float16* gh    = (_Float16*)(ws + 1 * SEG);
    float*    cell  = (float*)(ws + 1 * SEG);   // aliases gh (dead after gemm1)
    _Float16* ogin  = (_Float16*)(ws + 2 * SEG);
    float*    igbuf = (float*)(ws + 3 * SEG);   // igate, then igh in-place
    _Float16* hbuf  = (_Float16*)(ws + 4 * SEG);
    char*     wbase = ws + 6 * SEG;
    _Float16* wt1   = (_Float16*)(wbase);                         // [NH][768][256]
    _Float16* wt2   = (_Float16*)(wbase + 3145728ull);            // [NH][128][512]
    _Float16* wt3   = (_Float16*)(wbase + 3145728ull + 1048576ull); // [NH][128][256]
    float*    scanA = (float*)(wbase + 8388608ull);               // [32][16][128]
    float*    scanB = (float*)(wbase + 8388608ull + 262144ull);
    float*    carry = (float*)(wbase + 8388608ull + 524288ull);

    // Weight pre-transpose/convert (tiny; once per launch).
    wtrans_kernel<<<dim3(256 * 256 / 256, NH), 256, 0, stream>>>(W_ifg, wt1, 256, 256, 768, 0);
    wtrans_kernel<<<dim3(256 * 512 / 256, NH), 256, 0, stream>>>(W_h1,  wt1, 256, 512, 768, 256);
    wtrans_kernel<<<dim3(512 * 128 / 256, NH), 256, 0, stream>>>(W_h2,  wt2, 512, 128, 128, 0);
    wtrans_kernel<<<dim3(256 * 128 / 256, NH), 256, 0, stream>>>(W_og,  wt3, 256, 128, 128, 0);

    cumsum_kernel<<<BSZ * NH, AD, 0, stream>>>(x, csum);
    ln_build_kernel<<<NTOK, 256, 0, stream>>>(x, csum, ln_w, ln_b, gh, ogin);
    gemm1_kernel<<<dim3(NTOK / 128, 6, NH), 256, 0, stream>>>(
        gh, wt1, b_ifg, b_h1, fgate, igbuf, hbuf);
    gemm2_kernel<<<dim3(NTOK / 128, 1, NH), 256, 0, stream>>>(
        hbuf, wt2, b_h2, igbuf);
    scan_partial_kernel<<<BSZ * NH * NCH, AD, 0, stream>>>(fgate, igbuf, scanA, scanB);
    scan_carry_kernel<<<BSZ * NH, AD, 0, stream>>>(scanA, scanB, carry, init_cx);
    scan_apply_kernel<<<BSZ * NH * NCH, AD, 0, stream>>>(fgate, igbuf, carry, cell, ogin);
    gemm3_kernel<<<dim3(NTOK / 128, 1, NH), 256, 0, stream>>>(
        ogin, wt3, b_og, cell, out);
}
